// GNN_34162169872973
// MI455X (gfx1250) — matrix operand order repaired and checked by reading, not compiled, not run on hardware
//
#include <hip/hip_runtime.h>
#include <hip/hip_bf16.h>
#include <stdint.h>

typedef _Float16 v16h __attribute__((ext_vector_type(16)));
typedef _Float16 half8 __attribute__((ext_vector_type(8)));
typedef _Float16 half4 __attribute__((ext_vector_type(4)));
typedef float    v8f  __attribute__((ext_vector_type(8)));
typedef int      ivec4 __attribute__((vector_size(16)));   // matches builtin param type

#define NN 65536
#define EE 131072
#define DD 512
#define LL 5
#define GG 512

// ---------------- CDNA5 async global->LDS staging helpers ----------------

#if defined(__gfx1250__) && __has_builtin(__builtin_amdgcn_global_load_async_to_lds_b128)
#define HAS_ASYNC_LDS 1
#else
#define HAS_ASYNC_LDS 0
#endif

__device__ __forceinline__ void async_cp_b128(const _Float16* g, _Float16* l) {
#if HAS_ASYNC_LDS
    // generic->AS1: same 64-bit value; generic LDS ptr low 32 bits == LDS offset (AS3)
    __attribute__((address_space(1))) ivec4* gp =
        (__attribute__((address_space(1))) ivec4*)(uintptr_t)g;
    __attribute__((address_space(3))) ivec4* lp =
        (__attribute__((address_space(3))) ivec4*)(uint32_t)(uintptr_t)l;
    __builtin_amdgcn_global_load_async_to_lds_b128(gp, lp, 0, 0);
#else
    *(uint4*)l = *(const uint4*)g;   // sync fallback through VGPRs
#endif
}

__device__ __forceinline__ void wait_async_le4() {
#if HAS_ASYNC_LDS
#if __has_builtin(__builtin_amdgcn_s_wait_asynccnt)
    __builtin_amdgcn_s_wait_asynccnt(4);
#else
    asm volatile("s_wait_asynccnt 0x4" ::: "memory");
#endif
#endif
}

__device__ __forceinline__ void wait_async_le0() {
#if HAS_ASYNC_LDS
#if __has_builtin(__builtin_amdgcn_s_wait_asynccnt)
    __builtin_amdgcn_s_wait_asynccnt(0);
#else
    asm volatile("s_wait_asynccnt 0x0" ::: "memory");
#endif
#endif
}

// ---------------- elementwise / gather / scatter kernels ----------------

__global__ void k_embed(const int* __restrict__ x,
                        const float* __restrict__ xe1,
                        const float* __restrict__ xe2,
                        float* __restrict__ h) {
    int i = blockIdx.x * blockDim.x + threadIdx.x;   // over N*D/4
    int node = i >> 7;                 // 128 float4 per node
    int c = (i & 127) << 2;
    int a = x[node * 2 + 0];
    int ch = x[node * 2 + 1];
    float4 v1 = *(const float4*)(xe1 + a * DD + c);
    float4 v2 = *(const float4*)(xe2 + ch * DD + c);
    float4 o = make_float4(v1.x + v2.x, v1.y + v2.y, v1.z + v2.z, v1.w + v2.w);
    *(float4*)(h + (size_t)node * DD + c) = o;
}

__global__ void k_cvt_f32_f16(const float* __restrict__ in,
                              _Float16* __restrict__ out) {
    int i = blockIdx.x * blockDim.x + threadIdx.x;   // over n/4
    float4 v = ((const float4*)in)[i];
    half4 o;
    o.x = (_Float16)v.x; o.y = (_Float16)v.y;
    o.z = (_Float16)v.z; o.w = (_Float16)v.w;
    ((half4*)out)[i] = o;
}

// W [K, Nc] f32 row-major  ->  WT [Nc, K] f16 row-major (for contiguous B staging)
__global__ void k_cvt_w_t(const float* __restrict__ W, _Float16* __restrict__ WT,
                          int K, int Nc) {
    int i = blockIdx.x * blockDim.x + threadIdx.x;   // over K*Nc
    int n = i / K;
    int k = i - n * K;
    WT[i] = (_Float16)W[(size_t)k * Nc + n];
}

__global__ void k_message(const float* __restrict__ h,
                          const int* __restrict__ ei,
                          const int* __restrict__ ea,
                          const float* __restrict__ ee1,
                          const float* __restrict__ ee2,
                          float* __restrict__ agg) {
    int e = blockIdx.x;                 // E + N blocks
    int c = threadIdx.x << 2;           // 128 threads * 4 = 512 channels
    int src, dst, a0, a1;
    if (e < EE) {
        src = ei[e]; dst = ei[EE + e];
        a0 = ea[2 * e]; a1 = ea[2 * e + 1];
    } else {
        int v = e - EE;
        src = v; dst = v; a0 = 4; a1 = 0;   // self-loop attr [4,0]
    }
    float4 v1 = *(const float4*)(ee1 + a0 * DD + c);
    float4 v2 = *(const float4*)(ee2 + a1 * DD + c);
    float4 hv = *(const float4*)(h + (size_t)src * DD + c);
    float* p = agg + (size_t)dst * DD + c;
    atomicAdd(p + 0, hv.x + v1.x + v2.x);
    atomicAdd(p + 1, hv.y + v1.y + v2.y);
    atomicAdd(p + 2, hv.z + v1.z + v2.z);
    atomicAdd(p + 3, hv.w + v1.w + v2.w);
}

// ------- WMMA GEMM: C[M,Nc] = A[M,K](f16) * BT[Nc,K]^T(f16) + bias -------
// Double-buffered LDS, async global->LDS staging, 8 waves, wave tile 16x128.
// OUT_MODE 0: ReLU + store f16 ; OUT_MODE 1: store f32
#define BM 128
#define BN 128
#define BK 32
#define AST 48   // LDS row stride (halves): 96B, 16B multiple, bank-spreading pad

template<int OUT_MODE>
__global__ __launch_bounds__(256) void k_gemm(const _Float16* __restrict__ A,
                                              const _Float16* __restrict__ BT,
                                              const float* __restrict__ bias,
                                              _Float16* __restrict__ out16,
                                              float* __restrict__ out32,
                                              int K, int Nc) {
    __shared__ _Float16 As[2][BM * AST];
    __shared__ _Float16 Bs[2][BN * AST];
    const int tid  = threadIdx.x;
    const int wave = tid >> 5;
    const int lane = tid & 31;
    const int rowBase = blockIdx.y * BM;
    const int n0 = blockIdx.x * BN;

    const _Float16* gA = A + (size_t)rowBase * K;   // [BM rows][K]
    const _Float16* gB = BT + (size_t)n0 * K;       // [BN rows][K]

    v8f acc[8] = {};

    const int arow  = (wave << 4) + (lane & 15);
    const int koffA = (lane < 16) ? 0 : 16;   // halves
    const int koffB = (lane < 16) ? 0 : 16;   // halves
    const int lcol  = lane & 15;

    // stage one BK-slice of A and B into LDS buffer `sA`/`sB`
    auto stage = [&](int kbase, _Float16* sA, _Float16* sB) {
#pragma unroll
        for (int it = 0; it < 2; ++it) {
            int q = it * 256 + tid;           // 512 x 16B chunks per matrix
            int r = q >> 2;                   // row 0..127
            int c = (q & 3) << 3;             // half offset 0,8,16,24
            async_cp_b128(gA + (size_t)r * K + kbase + c, sA + r * AST + c);
            async_cp_b128(gB + (size_t)r * K + kbase + c, sB + r * AST + c);
        }
    };

    const int nk = K / BK;
    stage(0, As[0], Bs[0]);

    for (int i = 0; i < nk; ++i) {
        const int cur = i & 1;
        if (i + 1 < nk) {
            stage((i + 1) * BK, As[cur ^ 1], Bs[cur ^ 1]);
            wait_async_le4();     // current buffer's 4 async ops complete (in-order)
        } else {
            wait_async_le0();
        }
        __syncthreads();

        // A fragment: lanes<16 -> K 0..15; lanes>=16 -> K 16..31
        half8 lo = *(const half8*)&As[cur][arow * AST + koffA];
        half8 hi = *(const half8*)&As[cur][arow * AST + 8 + koffA];
        v16h a = __builtin_shufflevector(lo, hi,
                 0,1,2,3,4,5,6,7,8,9,10,11,12,13,14,15);
#pragma unroll
        for (int t = 0; t < 8; ++t) {
            // B fragment: lanes<16 -> col n, K 0..15; lanes>=16 -> col n, K 16..31
            int bcol = (t << 4) + lcol;
            v16h b = *(const v16h*)&Bs[cur][bcol * AST + koffB];
            acc[t] = __builtin_amdgcn_wmma_f32_16x16x32_f16(
                false, a, false, b, (short)0, acc[t], false, false);
        }
        __syncthreads();          // all waves done reading before buffer reuse
    }

    // epilogue: C/D layout -> VGPR i: lanes<16 row M=i, lanes>=16 row M=8+i, N=lane&15
    const int rsel = (lane < 16) ? 0 : 8;
#pragma unroll
    for (int t = 0; t < 8; ++t) {
        int col = n0 + (t << 4) + lcol;
        float bv = bias[col];
#pragma unroll
        for (int i = 0; i < 8; ++i) {
            int row = rowBase + (wave << 4) + rsel + i;
            float v = acc[t][i] + bv;
            if constexpr (OUT_MODE == 0) {
                v = fmaxf(v, 0.0f);
                out16[(size_t)row * Nc + col] = (_Float16)v;
            } else {
                out32[(size_t)row * Nc + col] = v;
            }
        }
    }
}

// ---------------- BatchNorm ----------------

__global__ void k_bn_stats(const float* __restrict__ x, float* __restrict__ stats) {
    int c2 = threadIdx.x;                 // float2 channel index
    int r0 = blockIdx.x * 256;
    float sx = 0.f, sy = 0.f, qx = 0.f, qy = 0.f;
    for (int r = 0; r < 256; ++r) {
        float2 v = *(const float2*)(x + (size_t)(r0 + r) * DD + (c2 << 1));
        sx += v.x; sy += v.y; qx += v.x * v.x; qy += v.y * v.y;
    }
    atomicAdd(&stats[(c2 << 1) + 0], sx);
    atomicAdd(&stats[(c2 << 1) + 1], sy);
    atomicAdd(&stats[DD + (c2 << 1) + 0], qx);
    atomicAdd(&stats[DD + (c2 << 1) + 1], qy);
}

__global__ void k_bn_finalize(const float* __restrict__ stats,
                              const float* __restrict__ gamma,
                              const float* __restrict__ beta,
                              float* __restrict__ scsh) {
    int c = threadIdx.x;                  // 512 threads
    float invn = 1.0f / (float)NN;
    float mean = stats[c] * invn;
    float var = stats[DD + c] * invn - mean * mean;
    float sc = gamma[c] * rsqrtf(var + 1e-5f);
    scsh[c] = sc;
    scsh[DD + c] = beta[c] - mean * sc;
}

__global__ void k_bn_apply(const float* __restrict__ x,
                           const float* __restrict__ scsh,
                           float* __restrict__ out, int relu) {
    int i = blockIdx.x * blockDim.x + threadIdx.x;   // over N*D/4
    int c = (i & 127) << 2;
    float4 v = ((const float4*)x)[i];
    float4 s = *(const float4*)(scsh + c);
    float4 b = *(const float4*)(scsh + DD + c);
    float4 o = make_float4(v.x * s.x + b.x, v.y * s.y + b.y,
                           v.z * s.z + b.z, v.w * s.w + b.w);
    if (relu) {
        o.x = fmaxf(o.x, 0.f); o.y = fmaxf(o.y, 0.f);
        o.z = fmaxf(o.z, 0.f); o.w = fmaxf(o.w, 0.f);
    }
    ((float4*)out)[i] = o;
}

// ---------------- graph pooling ----------------

__global__ void k_count(const int* __restrict__ batch, float* __restrict__ counts) {
    int i = blockIdx.x * blockDim.x + threadIdx.x;
    atomicAdd(&counts[batch[i]], 1.0f);
}

__global__ void k_pool(const float* __restrict__ h, const int* __restrict__ batch,
                       float* __restrict__ graph) {
    int node = blockIdx.x;
    int c = threadIdx.x << 2;
    int b = batch[node];
    float4 v = *(const float4*)(h + (size_t)node * DD + c);
    float* p = graph + (size_t)b * DD + c;
    atomicAdd(p + 0, v.x);
    atomicAdd(p + 1, v.y);
    atomicAdd(p + 2, v.z);
    atomicAdd(p + 3, v.w);
}

__global__ void k_pool_div(float* __restrict__ graph, const float* __restrict__ counts) {
    int i = blockIdx.x * blockDim.x + threadIdx.x;   // over G*D
    int g = i >> 9;
    graph[i] *= 1.0f / fmaxf(counts[g], 1.0f);
}

// ---------------- launch ----------------

extern "C" void kernel_launch(void* const* d_in, const int* in_sizes, int n_in,
                              void* d_out, int out_size, void* d_ws, size_t ws_size,
                              hipStream_t stream) {
    const int* x      = (const int*)d_in[0];
    const int* ei     = (const int*)d_in[1];
    const int* ea     = (const int*)d_in[2];
    const int* batch  = (const int*)d_in[3];
    const float* xe1  = (const float*)d_in[4];
    const float* xe2  = (const float*)d_in[5];
    const float* ee1  = (const float*)d_in[6];
    const float* ee2  = (const float*)d_in[7];
    const float* W1   = (const float*)d_in[8];
    const float* b1   = (const float*)d_in[9];
    const float* W2   = (const float*)d_in[10];
    const float* b2   = (const float*)d_in[11];
    const float* gam  = (const float*)d_in[12];
    const float* bet  = (const float*)d_in[13];

    char* ws = (char*)d_ws;
    size_t off = 0;
    float*    h     = (float*)(ws + off);  off += (size_t)NN * DD * 4;
    float*    agg   = (float*)(ws + off);  off += (size_t)NN * DD * 4;
    float*    hraw  = (float*)(ws + off);  off += (size_t)NN * DD * 4;
    _Float16* aggh  = (_Float16*)(ws + off); off += (size_t)NN * DD * 2;
    _Float16* hh1   = (_Float16*)(ws + off); off += (size_t)NN * 2 * DD * 2;
    _Float16* w1t   = (_Float16*)(ws + off); off += (size_t)DD * 2 * DD * 2;  // [1024][512]
    _Float16* w2t   = (_Float16*)(ws + off); off += (size_t)DD * 2 * DD * 2;  // [512][1024]
    float*    stats = (float*)(ws + off);  off += 2 * DD * 4;
    float*    scsh  = (float*)(ws + off);  off += 2 * DD * 4;
    float*    cnts  = (float*)(ws + off);  off += GG * 4;

    float* outGraph = (float*)d_out;
    float* outH     = outGraph + (size_t)GG * DD;

    const int ND4blocks = (NN * DD / 4) / 256;

    k_embed<<<ND4blocks, 256, 0, stream>>>(x, xe1, xe2, h);

    for (int l = 0; l < LL; ++l) {
        // weights -> f16, transposed to [Nc][K] for contiguous B staging
        k_cvt_w_t<<<(DD * 2 * DD) / 256, 256, 0, stream>>>(
            W1 + (size_t)l * DD * 2 * DD, w1t, DD, 2 * DD);
        k_cvt_w_t<<<(DD * 2 * DD) / 256, 256, 0, stream>>>(
            W2 + (size_t)l * 2 * DD * DD, w2t, 2 * DD, DD);

        // message passing (gather + edge emb + scatter-add, incl. self loops)
        (void)hipMemsetAsync(agg, 0, (size_t)NN * DD * 4, stream);
        k_message<<<EE + NN, 128, 0, stream>>>(h, ei, ea,
                                               ee1 + (size_t)l * 6 * DD,
                                               ee2 + (size_t)l * 3 * DD, agg);
        k_cvt_f32_f16<<<ND4blocks, 256, 0, stream>>>(agg, aggh);

        // GIN MLP: Linear(512,1024)+ReLU (f16 out), Linear(1024,512) (f32 out)
        k_gemm<0><<<dim3((2 * DD) / BN, NN / BM), 256, 0, stream>>>(
            aggh, w1t, b1 + (size_t)l * 2 * DD, hh1, nullptr, DD, 2 * DD);
        k_gemm<1><<<dim3(DD / BN, NN / BM), 256, 0, stream>>>(
            hh1, w2t, b2 + (size_t)l * DD, nullptr, hraw, 2 * DD, DD);

        // batchnorm (training-mode batch stats)
        (void)hipMemsetAsync(stats, 0, 2 * DD * 4, stream);
        k_bn_stats<<<NN / 256, 256, 0, stream>>>(hraw, stats);
        k_bn_finalize<<<1, DD, 0, stream>>>(stats, gam + (size_t)l * DD,
                                            bet + (size_t)l * DD, scsh);
        float* target = (l == LL - 1) ? outH : h;
        k_bn_apply<<<ND4blocks, 256, 0, stream>>>(hraw, scsh, target,
                                                  (l < LL - 1) ? 1 : 0);
    }

    // global mean pool over batch ids
    (void)hipMemsetAsync(cnts, 0, GG * 4, stream);
    (void)hipMemsetAsync(outGraph, 0, (size_t)GG * DD * 4, stream);
    k_count<<<NN / 256, 256, 0, stream>>>(batch, cnts);
    k_pool<<<NN, 128, 0, stream>>>(outH, batch, outGraph);
    k_pool_div<<<(GG * DD) / 256, 256, 0, stream>>>(outGraph, cnts);
}
